// MultiPersonKeypointModel_22857815949530
// MI455X (gfx1250) — compile-verified, matrix operand order checked by reading, not run on hardware
//
#include <hip/hip_runtime.h>
#include <math.h>

typedef __attribute__((ext_vector_type(16))) _Float16 v16h;
typedef __attribute__((ext_vector_type(8)))  float    v8f;

#define Bn    8
#define Cn    960
#define Hn    56
#define Wn    56
#define Pn    4
#define Kn    17
#define TOPKn 64
#define ROIn  56
#define REDn  60
#define HW    3136   // 56*56
#define Nroi  32     // B*P

__device__ __forceinline__ float sigmoidf(float x) { return 1.0f / (1.0f + expf(-x)); }

// ---------------------------------------------------------------------------
// Kernel 1: global avg + max pool over HxW per (b,c).  7680 blocks x 256 thr.
// ---------------------------------------------------------------------------
__global__ __launch_bounds__(256) void pool_kernel(
    const float* __restrict__ feat, float* __restrict__ avg, float* __restrict__ mx) {
  __shared__ float ssum[256];
  __shared__ float smax[256];
  const int bid = blockIdx.x;   // b*Cn + c
  const int tid = threadIdx.x;
  const float* p = feat + (size_t)bid * HW;
  float s = 0.0f, m = -INFINITY;
  for (int i = tid; i < HW; i += 256) { float v = p[i]; s += v; m = fmaxf(m, v); }
  ssum[tid] = s; smax[tid] = m; __syncthreads();
  for (int o = 128; o > 0; o >>= 1) {
    if (tid < o) { ssum[tid] += ssum[tid + o]; smax[tid] = fmaxf(smax[tid], smax[tid + o]); }
    __syncthreads();
  }
  if (tid == 0) { avg[bid] = ssum[0] * (1.0f / (float)HW); mx[bid] = smax[0]; }
}

// ---------------------------------------------------------------------------
// Kernel 2: SE MLP: scores = sigmoid(fc(avg)+fc(mx)).  8 blocks x 128 thr.
// ---------------------------------------------------------------------------
__global__ __launch_bounds__(128) void mlp_kernel(
    const float* __restrict__ avg, const float* __restrict__ mx,
    const float* __restrict__ fc1_w, const float* __restrict__ fc1_b,
    const float* __restrict__ fc2_w, const float* __restrict__ fc2_b,
    float* __restrict__ scores) {
  __shared__ float sA[Cn];
  __shared__ float sM[Cn];
  __shared__ float h1a[REDn];
  __shared__ float h1m[REDn];
  const int b = blockIdx.x, tid = threadIdx.x;
  for (int i = tid; i < Cn; i += 128) { sA[i] = avg[b * Cn + i]; sM[i] = mx[b * Cn + i]; }
  __syncthreads();
  if (tid < REDn) {
    float s = fc1_b[tid];
    const float* wrow = fc1_w + tid * Cn;
    for (int c = 0; c < Cn; ++c) s += wrow[c] * sA[c];
    h1a[tid] = fmaxf(s, 0.0f);
  } else if (tid >= 64 && tid < 64 + REDn) {
    int r = tid - 64;
    float s = fc1_b[r];
    const float* wrow = fc1_w + r * Cn;
    for (int c = 0; c < Cn; ++c) s += wrow[c] * sM[c];
    h1m[r] = fmaxf(s, 0.0f);
  }
  __syncthreads();
  for (int o = tid; o < Cn; o += 128) {
    float s = 2.0f * fc2_b[o];
    const float* wrow = fc2_w + o * REDn;
    for (int r = 0; r < REDn; ++r) s += wrow[r] * (h1a[r] + h1m[r]);
    scores[b * Cn + o] = sigmoidf(s);
  }
}

// ---------------------------------------------------------------------------
// Kernel 3: exact top-64 (descending, lowest-index ties) per batch.
// 8 blocks x 256 thr, 64 iterations of block-argmax.
// ---------------------------------------------------------------------------
__global__ __launch_bounds__(256) void topk_kernel(
    const float* __restrict__ scores, int* __restrict__ topidx) {
  __shared__ float sc[Cn];
  __shared__ float rv[256];
  __shared__ int   ri[256];
  const int b = blockIdx.x, tid = threadIdx.x;
  for (int i = tid; i < Cn; i += 256) sc[i] = scores[b * Cn + i];
  __syncthreads();
  for (int it = 0; it < TOPKn; ++it) {
    float bv = -INFINITY; int bi = 0x7fffffff;
    for (int i = tid; i < Cn; i += 256) {
      float v = sc[i];
      if (v > bv || (v == bv && i < bi)) { bv = v; bi = i; }
    }
    rv[tid] = bv; ri[tid] = bi; __syncthreads();
    for (int o = 128; o > 0; o >>= 1) {
      if (tid < o) {
        float v2 = rv[tid + o]; int i2 = ri[tid + o];
        if (v2 > rv[tid] || (v2 == rv[tid] && i2 < ri[tid])) { rv[tid] = v2; ri[tid] = i2; }
      }
      __syncthreads();
    }
    if (tid == 0) { topidx[b * TOPKn + it] = ri[0]; sc[ri[0]] = -INFINITY; }
    __syncthreads();
  }
}

// ---------------------------------------------------------------------------
// Kernel 4: fused ROI-align + 64->17 head matmul (WMMA f32.16x16x32.f16)
//           + sigmoid -> heat [32,17,56,56].
// grid = (32 rois, 7), block = 256 (8 waves of 32). Each wave owns 16-pixel
// M-tiles; K = 64 selected channels (2 WMMA K-steps); N = 17 (2 N-tiles).
// ---------------------------------------------------------------------------
__global__ __launch_bounds__(256) void roi_head_kernel(
    const float* __restrict__ feat, const float* __restrict__ bboxes,
    const int* __restrict__ topidx, const float* __restrict__ head_w,
    const float* __restrict__ head_b, float* __restrict__ heat) {
  __shared__ int chanoff[TOPKn];   // (b*Cn + selected_channel)*HW
  const int n = blockIdx.x;        // roi id = b*P + p
  const int b = n >> 2;

  // box -> pixel-space ROI (torchvision roi_align, sampling_ratio=1)
  const float cx = bboxes[n * 4 + 0], cy = bboxes[n * 4 + 1];
  const float bw = bboxes[n * 4 + 2], bh = bboxes[n * 4 + 3];
  const float bx1 = fminf(fmaxf(cx - 0.5f * bw, 0.0f), 1.0f) * (float)Wn;
  const float by1 = fminf(fmaxf(cy - 0.5f * bh, 0.0f), 1.0f) * (float)Hn;
  const float bx2 = fmaxf(fminf(fmaxf(cx + 0.5f * bw, 0.0f), 1.0f) * (float)Wn, bx1 + 1.0f);
  const float by2 = fmaxf(fminf(fmaxf(cy + 0.5f * bh, 0.0f), 1.0f) * (float)Hn, by1 + 1.0f);
  const float bwp = (bx2 - bx1) * (1.0f / (float)ROIn);
  const float bhp = (by2 - by1) * (1.0f / (float)ROIn);

  if (threadIdx.x < TOPKn)
    chanoff[threadIdx.x] = (b * Cn + topidx[b * TOPKn + threadIdx.x]) * HW;
  __syncthreads();

  const int lane   = threadIdx.x & 31;
  const int wid    = threadIdx.x >> 5;
  const int half   = lane >> 4;   // K-half selector (ISA 16-bit A/B striping)
  const int within = lane & 15;   // M (for A) / N (for B / C / D)

  // K index map per ISA 16-bit A-matrix layout: elem e -> K within a 32-chunk
  int km[16];
#pragma unroll
  for (int e = 0; e < 16; ++e) {
    int vg = e >> 1, pos = e & 1;
    km[e] = (vg < 4) ? (half * 8 + vg * 2 + pos) : (16 + half * 8 + (vg - 4) * 2 + pos);
  }

  // Build constant B fragments from head_w[c][k] (c = WMMA K dim, k = N dim)
  v16h bf[2][2];
#pragma unroll
  for (int j = 0; j < 2; ++j) {
    const int kout = j * 16 + within;
#pragma unroll
    for (int s = 0; s < 2; ++s) {
      v16h t;
#pragma unroll
      for (int e = 0; e < 16; ++e) {
        int c = s * 32 + km[e];
        float v = (kout < Kn) ? head_w[c * Kn + kout] : 0.0f;
        t[e] = (_Float16)v;
      }
      bf[j][s] = t;
    }
  }

  // wave-uniform tile loop (EXEC all-ones at every WMMA)
  for (int t = blockIdx.y * 8 + wid; t < HW / 16; t += 56) {
    const int p  = t * 16 + within;  // this lane's pixel (A row M=within)
    const int py = p / Wn, px = p % Wn;
    float xs = fminf(fmaxf(bx1 + ((float)px + 0.5f) * bwp, 0.0f), (float)(Wn - 1));
    float ys = fminf(fmaxf(by1 + ((float)py + 0.5f) * bhp, 0.0f), (float)(Hn - 1));
    int x0 = (int)floorf(xs), y0 = (int)floorf(ys);
    int x1i = min(x0 + 1, Wn - 1), y1i = min(y0 + 1, Hn - 1);
    float fx = xs - (float)x0, fy = ys - (float)y0;
    float w00 = (1.0f - fy) * (1.0f - fx), w01 = (1.0f - fy) * fx;
    float w10 = fy * (1.0f - fx),          w11 = fy * fx;
    int o00 = y0 * Wn + x0, o01 = y0 * Wn + x1i;
    int o10 = y1i * Wn + x0, o11 = y1i * Wn + x1i;

    v16h a0, a1;
#pragma unroll
    for (int e = 0; e < 16; ++e) {
      const int K = km[e];
      {
        const float* fp = feat + chanoff[K];
        a0[e] = (_Float16)(w00 * fp[o00] + w01 * fp[o01] + w10 * fp[o10] + w11 * fp[o11]);
      }
      {
        const float* fp = feat + chanoff[32 + K];
        a1[e] = (_Float16)(w00 * fp[o00] + w01 * fp[o01] + w10 * fp[o10] + w11 * fp[o11]);
      }
    }

#pragma unroll
    for (int j = 0; j < 2; ++j) {
      v8f acc = {};
      acc = __builtin_amdgcn_wmma_f32_16x16x32_f16(false, a0, false, bf[j][0],
                                                   (short)0, acc, false, false);
      acc = __builtin_amdgcn_wmma_f32_16x16x32_f16(false, a1, false, bf[j][1],
                                                   (short)0, acc, false, false);
      const int kout = j * 16 + within;   // D: lane -> N column
      if (kout < Kn) {
        const float hb = head_b[kout];
#pragma unroll
        for (int r = 0; r < 8; ++r) {
          const int M = r + 8 * half;     // D: vgpr r -> row M (+8 for upper half)
          const int pout = t * 16 + M;
          heat[((size_t)n * Kn + kout) * HW + pout] = sigmoidf(acc[r] + hb);
        }
      }
    }
  }
}

// ---------------------------------------------------------------------------
// Kernel 5: decode heat -> keypoints + vis.  544 blocks (n*17+k) x 256 thr.
// ---------------------------------------------------------------------------
__global__ __launch_bounds__(256) void decode_kernel(
    const float* __restrict__ heat, const float* __restrict__ bboxes,
    float* __restrict__ kpts, float* __restrict__ visout) {
  __shared__ float rv[256];
  __shared__ int   ri[256];
  const int mapi = blockIdx.x;          // n*Kn + k
  const int n = mapi / Kn;
  const int tid = threadIdx.x;
  const float* hm = heat + (size_t)mapi * HW;
  float bv = -INFINITY; int bi = 0x7fffffff;
  for (int i = tid; i < HW; i += 256) {
    float v = hm[i];
    if (v > bv || (v == bv && i < bi)) { bv = v; bi = i; }
  }
  rv[tid] = bv; ri[tid] = bi; __syncthreads();
  for (int o = 128; o > 0; o >>= 1) {
    if (tid < o) {
      float v2 = rv[tid + o]; int i2 = ri[tid + o];
      if (v2 > rv[tid] || (v2 == rv[tid] && i2 < ri[tid])) { rv[tid] = v2; ri[tid] = i2; }
    }
    __syncthreads();
  }
  if (tid == 0) {
    const float vis = rv[0];
    const int idx = ri[0];
    const int y = idx / Wn, x = idx % Wn;
    float wsum = 0.0f, ysum = 0.0f, xsum = 0.0f;
    for (int dy = -1; dy <= 1; ++dy)
      for (int dx = -1; dx <= 1; ++dx) {
        int yy = y + dy, xx = x + dx;
        if (yy >= 0 && yy < Hn && xx >= 0 && xx < Wn) {
          float v = hm[yy * Wn + xx];
          wsum += v; ysum += (float)yy * v; xsum += (float)xx * v;
        }
      }
    float yref, xref;
    if (wsum > 0.0f) {
      float inv = 1.0f / fmaxf(wsum, 1e-12f);
      yref = ysum * inv; xref = xsum * inv;
    } else {
      yref = (float)y; xref = (float)x;
    }
    yref *= 1.0f / (float)(Hn - 1);
    xref *= 1.0f / (float)(Wn - 1);
    const float cx = bboxes[n * 4 + 0], cy = bboxes[n * 4 + 1];
    const float bw = bboxes[n * 4 + 2], bh = bboxes[n * 4 + 3];
    const float kx = fminf(fmaxf(xref * bw + (cx - 0.5f * bw), 0.0f), 1.0f);
    const float ky = fminf(fmaxf(yref * bh + (cy - 0.5f * bh), 0.0f), 1.0f);
    kpts[mapi * 2 + 0] = kx;
    kpts[mapi * 2 + 1] = ky;
    visout[mapi] = vis;
  }
}

// ---------------------------------------------------------------------------
extern "C" void kernel_launch(void* const* d_in, const int* in_sizes, int n_in,
                              void* d_out, int out_size, void* d_ws, size_t ws_size,
                              hipStream_t stream) {
  const float* feat   = (const float*)d_in[0];
  const float* bboxes = (const float*)d_in[1];
  const float* fc1_w  = (const float*)d_in[2];
  const float* fc1_b  = (const float*)d_in[3];
  const float* fc2_w  = (const float*)d_in[4];
  const float* fc2_b  = (const float*)d_in[5];
  const float* head_w = (const float*)d_in[6];
  const float* head_b = (const float*)d_in[7];

  float* out = (float*)d_out;
  // outputs concatenated flat: keypoints[8,4,17,2], vis[8,4,17], heat[8,4,17,56,56]
  float* kpts = out;                 // 1088 floats
  float* vis  = out + 1088;          // 544 floats
  float* heat = out + 1632;          // 1,705,984 floats

  float* ws     = (float*)d_ws;
  float* avg    = ws;                // 7680
  float* mx     = ws + 7680;         // 7680
  float* scores = ws + 15360;        // 7680
  int*   topidx = (int*)(ws + 23040);// 512 ints

  pool_kernel<<<Bn * Cn, 256, 0, stream>>>(feat, avg, mx);
  mlp_kernel<<<Bn, 128, 0, stream>>>(avg, mx, fc1_w, fc1_b, fc2_w, fc2_b, scores);
  topk_kernel<<<Bn, 256, 0, stream>>>(scores, topidx);
  roi_head_kernel<<<dim3(Nroi, 7), 256, 0, stream>>>(feat, bboxes, topidx, head_w, head_b, heat);
  decode_kernel<<<Nroi * Kn, 256, 0, stream>>>(heat, bboxes, kpts, vis);
}
